// GNNLayer_35648228556980
// MI455X (gfx1250) — matrix operand order repaired and checked by reading, not compiled, not run on hardware
//
#include <hip/hip_runtime.h>
#include <math.h>

typedef __attribute__((ext_vector_type(16))) _Float16 v16h;
typedef __attribute__((ext_vector_type(8)))  _Float16 v8h;
typedef __attribute__((ext_vector_type(8)))  float    v8f;
typedef __attribute__((ext_vector_type(4)))  float    v4f;

#define DIM 128

static const int NNODE[6] = {20000, 100000, 4000, 8000, 10000, 4000};
static const int NOFF[6]  = {0, 20000, 120000, 124000, 132000, 142000};
static const int TOTAL_NODES = 146000;
static const int NREL = 10;
static const int RSRC[10]  = {0, 1, 1, 2, 1, 3, 1, 4, 1, 5};
static const int RDST[10]  = {1, 0, 2, 1, 3, 1, 4, 1, 5, 1};
static const int NEDGE[10] = {100000, 100000, 300000, 300000, 200000,
                              200000, 250000, 250000, 300000, 300000};
static const int NLAYERS = 4;
static const int E_LBL = 200000;

// ---------------------------------------------------------------------------
// utility kernels
// ---------------------------------------------------------------------------
__global__ void fill_zero(float* __restrict__ p, long long n) {
    long long i = (long long)blockIdx.x * blockDim.x + threadIdx.x;
    long long stride = (long long)gridDim.x * blockDim.x;
    for (; i < n; i += stride) p[i] = 0.0f;
}

// f32 -> f16 convert, optional fused ReLU (layer epilogue / input embedding init)
__global__ void convert_f16(const float* __restrict__ src,
                            _Float16* __restrict__ dst, long long n, int doRelu) {
    long long i = (long long)blockIdx.x * blockDim.x + threadIdx.x;
    long long stride = (long long)gridDim.x * blockDim.x;
    for (; i < n; i += stride) {
        float v = src[i];
        if (doRelu) v = fmaxf(v, 0.0f);
        dst[i] = (_Float16)v;
    }
}

// Swizzle fp32 weights into fragment-major packed f16:
//   Wh[ m*K*128 + ((kc*8+nt)*32 + lane)*16 + i ] = W[m][kc*32+kb+i][nt*16+n]
// where n=lane&15, kb = (lane&16)?8:0, i>=8 maps to k+8.  One B-fragment = 16 contiguous halves.
__global__ void swizzle_weights(const float* __restrict__ W,
                                _Float16* __restrict__ Wh, int kdim,
                                long long total) {
    long long per = (long long)kdim * DIM;
    long long t = (long long)blockIdx.x * blockDim.x + threadIdx.x;
    long long stride = (long long)gridDim.x * blockDim.x;
    for (; t < total; t += stride) {
        long long m = t / per;
        long long s = t % per;
        int i    = (int)(s & 15);
        int lane = (int)((s >> 4) & 31);
        int nt   = (int)((s >> 9) & 7);
        int kc   = (int)(s >> 12);
        int n  = lane & 15;
        int kb = (lane & 16) ? 8 : 0;
        int k  = kc * 32 + kb + (i < 8 ? i : i + 8);
        int col = nt * 16 + n;
        Wh[m * per + s] = (_Float16)W[m * per + (long long)k * DIM + col];
    }
}

// scatter: mean[dst] += (float)xh[src]; deg[dst] += 1  (one wave/edge, 4 el/lane)
__global__ void scatter_sum(const int* __restrict__ ei, int E,
                            const _Float16* __restrict__ xh_src,
                            float* __restrict__ meanbuf,
                            float* __restrict__ deg) {
    int t = blockIdx.x * blockDim.x + threadIdx.x;
    int e = t >> 5;
    if (e >= E) return;
    int lane = t & 31;
    int s  = ei[e];
    int d2 = ei[E + e];
    const _Float16* sp = xh_src + (long long)s * DIM + lane * 4;
    float* mp = meanbuf + (long long)d2 * DIM + lane * 4;
    atomicAdd(&mp[0], (float)sp[0]);
    atomicAdd(&mp[1], (float)sp[1]);
    atomicAdd(&mp[2], (float)sp[2]);
    atomicAdd(&mp[3], (float)sp[3]);
    if (lane == 0) atomicAdd(&deg[d2], 1.0f);
}

// mean finalize: divide by max(deg,1) and convert to f16 in one pass
__global__ void finalize_mean(const float* __restrict__ mean,
                              const float* __restrict__ deg,
                              _Float16* __restrict__ meanh, long long n) {
    long long i = (long long)blockIdx.x * blockDim.x + threadIdx.x;
    long long stride = (long long)gridDim.x * blockDim.x;
    for (; i < n; i += stride) {
        int row = (int)(i >> 7);
        float inv = 1.0f / fmaxf(deg[row], 1.0f);
        meanh[i] = (_Float16)(mean[i] * inv);
    }
}

// ---------------------------------------------------------------------------
// fragment loaders (all packed f16, no conversions)
// A 16x32: lane row = lane&15; lanes 0-15 K={0..7,16..23}, 16-31 K={8..15,24..31}
// ---------------------------------------------------------------------------
__device__ inline v16h load_a_frag_h(const _Float16* __restrict__ base, int row,
                                     int kbase) {
    int koff = (threadIdx.x & 16) ? 8 : 0;
    const _Float16* p = base + (long long)row * DIM + kbase + koff;
    v8h lo = *(const v8h*)p;           // global_load_b128
    v8h hi = *(const v8h*)(p + 16);    // global_load_b128
    return __builtin_shufflevector(lo, hi, 0, 1, 2, 3, 4, 5, 6, 7,
                                   8, 9, 10, 11, 12, 13, 14, 15);
}

__device__ inline v16h load_b_swz(const _Float16* Wswz, int kc, int nt) {
    int lane = threadIdx.x & 31;
    return *(const v16h*)(Wswz + (((kc * 8 + nt) * 32) + lane) * 16);
}

// ---------------------------------------------------------------------------
// Fused hetero-SAGE relation kernel (4 waves / 64 rows per block):
//   out[dst,:] += meanh[dst,:] @ Wl + bl + xh[dst,:] @ Wr
// Weights staged in LDS; accumulator tile routed through LDS so the global
// accumulation is wide b128 RMW with wave-uniform row guards (no divergence).
// ---------------------------------------------------------------------------
__global__ void __launch_bounds__(128)
sage_gemm_fused(const _Float16* __restrict__ meanh,
                const _Float16* __restrict__ xh_dst,
                const _Float16* __restrict__ Wlh, const float* __restrict__ bl,
                const _Float16* __restrict__ Wrh,
                float* __restrict__ out, int ndst) {
    extern __shared__ _Float16 smem[];        // 64KB weights + 32KB f32 tiles
    _Float16* sWl = smem;                     // 16384 halves
    _Float16* sWr = smem + 16384;             // 16384 halves
    float* sTile = (float*)(smem + 32768);    // 4 waves * 16 rows * 128 f32

    __builtin_prefetch(Wlh, 0, 1);            // global_prefetch_b8
    __builtin_prefetch(Wrh, 0, 1);

    {   // cooperative stage of both 128x128 f16 weight matrices into LDS
        const v8h* gl = (const v8h*)Wlh;
        const v8h* gr = (const v8h*)Wrh;
        v8h* sl = (v8h*)sWl;
        v8h* sr = (v8h*)sWr;
        for (int i = threadIdx.x; i < 2048; i += 128) {
            sl[i] = gl[i];
            sr[i] = gr[i];
        }
    }
    __syncthreads();

    int wave = threadIdx.x >> 5;
    int lane = threadIdx.x & 31;
    int rowbase = blockIdx.x * 64 + wave * 16;
    if (rowbase >= ndst) return;
    int r = rowbase + (lane & 15);
    if (r >= ndst) r = ndst - 1;
    float* sOut = sTile + wave * 16 * DIM;

    v16h am[4], ax[4];
#pragma unroll
    for (int kc = 0; kc < 4; ++kc) {
        am[kc] = load_a_frag_h(meanh,  r, kc * 32);
        ax[kc] = load_a_frag_h(xh_dst, r, kc * 32);
    }

    int n  = lane & 15;
    int mh = (lane >> 4) * 8;
#pragma unroll
    for (int nt = 0; nt < 8; ++nt) {
        v8f c = {};
#pragma unroll
        for (int kc = 0; kc < 4; ++kc) {
            v16h bL = load_b_swz(sWl, kc, nt);
            c = __builtin_amdgcn_wmma_f32_16x16x32_f16(false, am[kc], false, bL,
                                                       (short)0, c, false, false);
            v16h bR = load_b_swz(sWr, kc, nt);
            c = __builtin_amdgcn_wmma_f32_16x16x32_f16(false, ax[kc], false, bR,
                                                       (short)0, c, false, false);
        }
        float bias = bl[nt * 16 + n];
        // park the 16x16 tile in LDS (restores row-major contiguity)
#pragma unroll
        for (int j = 0; j < 8; ++j)
            sOut[(mh + j) * DIM + nt * 16 + n] = c[j] + bias;
    }

    // wave-uniform guarded accumulation: one full row per iteration
#pragma unroll 4
    for (int it = 0; it < 16; ++it) {
        int grow = rowbase + it;
        if (grow < ndst) {   // uniform across the wave -> scalar branch
            float* gp = out + (long long)grow * DIM + lane * 4;
            const float* sp = sOut + it * DIM + lane * 4;
            v4f g = *(const v4f*)gp;   // global_load_b128
            v4f s = *(const v4f*)sp;   // ds_load_b128
            g += s;
            *(v4f*)gp = g;             // global_store_b128
        }
    }
}

// ---------------------------------------------------------------------------
// Fused classifier, per 16 labeled edges (one wave):
//   h = relu([zv|zd] @ W1 + b1)  (WMMA, K=256, w1 pre-swizzled f16)
//   pred = h @ w2 + b2           (folded, shfl reduction)
// ---------------------------------------------------------------------------
__global__ void __launch_bounds__(32)
classifier_kernel(const _Float16* __restrict__ xv, const _Float16* __restrict__ xd,
                  const int* __restrict__ eli, const _Float16* __restrict__ w1h,
                  const float* __restrict__ b1, const float* __restrict__ w2,
                  const float* __restrict__ b2, float* __restrict__ pred, int E) {
    int lane = threadIdx.x;
    int base = blockIdx.x * 16;
    int er = base + (lane & 15);
    if (er >= E) er = E - 1;
    int vr = eli[er];
    int dr = eli[E + er];

    __builtin_prefetch(w1h, 0, 1);

    v16h a[8];
#pragma unroll
    for (int kc = 0; kc < 4; ++kc) a[kc]     = load_a_frag_h(xv, vr, kc * 32);
#pragma unroll
    for (int kc = 0; kc < 4; ++kc) a[4 + kc] = load_a_frag_h(xd, dr, kc * 32);

    float acc[8];
#pragma unroll
    for (int j = 0; j < 8; ++j) acc[j] = 0.0f;
    int n = lane & 15;

#pragma unroll
    for (int nt = 0; nt < 8; ++nt) {
        v8f c = {};
#pragma unroll
        for (int kc = 0; kc < 8; ++kc) {
            v16h b = load_b_swz(w1h, kc, nt);
            c = __builtin_amdgcn_wmma_f32_16x16x32_f16(false, a[kc], false, b,
                                                       (short)0, c, false, false);
        }
        float bias = b1[nt * 16 + n];
        float w2v  = w2[nt * 16 + n];
#pragma unroll
        for (int j = 0; j < 8; ++j) {
            float h = fmaxf(c[j] + bias, 0.0f);
            acc[j] += h * w2v;
        }
    }
#pragma unroll
    for (int j = 0; j < 8; ++j) {
        acc[j] += __shfl_xor(acc[j], 1, 16);
        acc[j] += __shfl_xor(acc[j], 2, 16);
        acc[j] += __shfl_xor(acc[j], 4, 16);
        acc[j] += __shfl_xor(acc[j], 8, 16);
    }
    if ((lane & 15) == 0) {
        int mh = (lane >> 4) * 8;
        float b2v = b2[0];
#pragma unroll
        for (int j = 0; j < 8; ++j) {
            int row = base + mh + j;
            if (row < E) pred[row] = acc[j] + b2v;
        }
    }
}

// ---------------------------------------------------------------------------
// BCE-with-logits loss reduction
// ---------------------------------------------------------------------------
__global__ void loss_partial(const float* __restrict__ pred,
                             const float* __restrict__ label,
                             float* __restrict__ acc, int E) {
    __shared__ float sdata[256];
    int tid = threadIdx.x;
    int i = blockIdx.x * blockDim.x + tid;
    float v = 0.0f;
    if (i < E) {
        float p = pred[i];
        float y = label[i];
        float sp = fmaxf(p, 0.0f) + log1pf(expf(-fabsf(p)));  // stable softplus
        v = sp - p * y;
    }
    sdata[tid] = v;
    __syncthreads();
    for (int s = 128; s > 0; s >>= 1) {
        if (tid < s) sdata[tid] += sdata[tid + s];
        __syncthreads();
    }
    if (tid == 0) atomicAdd(acc, sdata[0]);
}

__global__ void loss_final(const float* __restrict__ acc, float* __restrict__ out,
                           float invE) {
    out[0] = acc[0] * invE;
}

// ---------------------------------------------------------------------------
// host driver
// ---------------------------------------------------------------------------
extern "C" void kernel_launch(void* const* d_in, const int* in_sizes, int n_in,
                              void* d_out, int out_size, void* d_ws, size_t ws_size,
                              hipStream_t stream) {
    (void)in_sizes; (void)n_in; (void)out_size; (void)ws_size;

    const int* ei[NREL];
    for (int r = 0; r < NREL; ++r) ei[r] = (const int*)d_in[12 + r];
    const float* Wl = (const float*)d_in[22];
    const float* bl = (const float*)d_in[23];
    const float* Wr = (const float*)d_in[24];
    const float* w1 = (const float*)d_in[25];
    const float* b1 = (const float*)d_in[26];
    const float* w2 = (const float*)d_in[27];
    const float* b2 = (const float*)d_in[28];
    const int* eli = (const int*)d_in[29];
    const float* elab = (const float*)d_in[30];
    float* out = (float*)d_out;  // out[0] = loss, out[1..E] = pred

    char* ws = (char*)d_ws;
    size_t off = 0;
    auto alloc = [&](size_t bytes) -> void* {
        void* p = ws + off;
        off = (off + bytes + 255) & ~(size_t)255;
        return p;
    };
    _Float16* xh    = (_Float16*)alloc((size_t)TOTAL_NODES * DIM * 2);
    float*    xn    = (float*)alloc((size_t)TOTAL_NODES * DIM * 4);
    float*    mean  = (float*)alloc((size_t)100000 * DIM * 4);
    _Float16* meanh = (_Float16*)alloc((size_t)100000 * DIM * 2);
    float*    deg   = (float*)alloc((size_t)100000 * 4);
    _Float16* Wlh   = (_Float16*)alloc((size_t)NLAYERS * NREL * DIM * DIM * 2);
    _Float16* Wrh   = (_Float16*)alloc((size_t)NLAYERS * NREL * DIM * DIM * 2);
    _Float16* w1h   = (_Float16*)alloc((size_t)2 * DIM * DIM * 2);
    float*    lacc  = (float*)alloc(4);

    // one-time weight swizzle+convert to fragment-major f16
    {
        long long tw = (long long)NLAYERS * NREL * DIM * DIM;
        swizzle_weights<<<(int)((tw + 255) / 256), 256, 0, stream>>>(Wl, Wlh, DIM, tw);
        swizzle_weights<<<(int)((tw + 255) / 256), 256, 0, stream>>>(Wr, Wrh, DIM, tw);
        long long tc = (long long)2 * DIM * DIM;
        swizzle_weights<<<(int)((tc + 255) / 256), 256, 0, stream>>>(w1, w1h, 2 * DIM, tc);
    }

    // node_id_* are arange -> feature init = convert embeddings to f16
    for (int t = 0; t < 6; ++t) {
        long long n = (long long)NNODE[t] * DIM;
        convert_f16<<<(int)((n + 255) / 256), 256, 0, stream>>>(
            (const float*)d_in[6 + t], xh + (size_t)NOFF[t] * DIM, n, 0);
    }

    long long nxn = (long long)TOTAL_NODES * DIM;
    for (int l = 0; l < NLAYERS; ++l) {
        fill_zero<<<(int)((nxn + 255) / 256), 256, 0, stream>>>(xn, nxn);
        for (int r = 0; r < NREL; ++r) {
            int ndst = NNODE[RDST[r]];
            long long nm = (long long)ndst * DIM;
            fill_zero<<<(int)((nm + 255) / 256), 256, 0, stream>>>(mean, nm);
            fill_zero<<<(ndst + 255) / 256, 256, 0, stream>>>(deg, ndst);
            int E = NEDGE[r];
            const _Float16* xsrc = xh + (size_t)NOFF[RSRC[r]] * DIM;
            long long nth = (long long)E * 32;
            scatter_sum<<<(int)((nth + 255) / 256), 256, 0, stream>>>(
                ei[r], E, xsrc, mean, deg);
            finalize_mean<<<(int)((nm + 255) / 256), 256, 0, stream>>>(
                mean, deg, meanh, nm);

            const _Float16* Wlp = Wlh + ((size_t)l * NREL + r) * DIM * DIM;
            const float*    blp = bl  + ((size_t)l * NREL + r) * DIM;
            const _Float16* Wrp = Wrh + ((size_t)l * NREL + r) * DIM * DIM;
            const _Float16* xdst = xh + (size_t)NOFF[RDST[r]] * DIM;
            float* outp = xn + (size_t)NOFF[RDST[r]] * DIM;
            sage_gemm_fused<<<(ndst + 63) / 64, 128, 98304, stream>>>(
                meanh, xdst, Wlp, blp, Wrp, outp, ndst);
        }
        // epilogue: (ReLU for layers 0..2) + convert back to f16 features
        convert_f16<<<(int)((nxn + 255) / 256), 256, 0, stream>>>(
            xn, xh, nxn, (l < NLAYERS - 1) ? 1 : 0);
    }

    const _Float16* xv = xh + (size_t)NOFF[1] * DIM;
    const _Float16* xd = xh + (size_t)NOFF[5] * DIM;
    classifier_kernel<<<(E_LBL + 15) / 16, 32, 0, stream>>>(
        xv, xd, eli, w1h, b1, w2, b2, out + 1, E_LBL);

    fill_zero<<<1, 256, 0, stream>>>(lacc, 1);
    loss_partial<<<(E_LBL + 255) / 256, 256, 0, stream>>>(out + 1, elab, lacc, E_LBL);
    loss_final<<<1, 1, 0, stream>>>(lacc, out, 1.0f / (float)E_LBL);
}
